// EVDLoRA_83794811945494
// MI455X (gfx1250) — matrix operand order repaired and checked by reading, not compiled, not run on hardware
//
#include <hip/hip_runtime.h>

typedef __attribute__((ext_vector_type(4)))  __bf16 v4bf;
typedef __attribute__((ext_vector_type(8)))  __bf16 v8bf;
typedef __attribute__((ext_vector_type(16))) __bf16 v16bf;
typedef __attribute__((ext_vector_type(8)))  float  v8f;
typedef __attribute__((ext_vector_type(4)))  int    v4i;

typedef __attribute__((address_space(1))) v4i* as1_v4i_p;
typedef __attribute__((address_space(3))) v4i* as3_v4i_p;

#if __has_builtin(__builtin_amdgcn_global_load_async_to_lds_b128)
#define EVD_HAS_ASYNC_LDS 1
#else
#define EVD_HAS_ASYNC_LDS 0
#endif

__global__ void evd_zero_out(float* out) { out[0] = 0.0f; }

// ---------- bf16 hi/lo split helpers ----------
__device__ __forceinline__ void split4(float4 f, v4bf* hi, v4bf* lo) {
    float xs[4] = {f.x, f.y, f.z, f.w};
    #pragma unroll
    for (int e = 0; e < 4; ++e) {
        __bf16 h = (__bf16)xs[e];
        (*hi)[e] = h;
        (*lo)[e] = (__bf16)(xs[e] - (float)h);
    }
}

// One-shot preprocess: M[8192x256] f32 -> Whi/Wlo bf16 (error-compensated split).
__global__ void evd_split_bf16(const float* __restrict__ A,
                               __bf16* __restrict__ whi, __bf16* __restrict__ wlo) {
    int idx = blockIdx.x * blockDim.x + threadIdx.x;   // float4 index, 524288 total
    float4 f = *(const float4*)&A[(size_t)idx * 4];
    v4bf hi, lo;
    split4(f, &hi, &lo);
    *(v4bf*)&whi[(size_t)idx * 4] = hi;
    *(v4bf*)&wlo[(size_t)idx * 4] = lo;
}

// ---------- async global->LDS copy (16B per lane) ----------
__device__ __forceinline__ void async_copy16(const __bf16* gsrc, __bf16* ldst) {
#if EVD_HAS_ASYNC_LDS
    __builtin_amdgcn_global_load_async_to_lds_b128(
        (as1_v4i_p)(void*)gsrc,
        (as3_v4i_p)(void*)ldst,
        0, 0);
#else
    *(v8bf*)ldst = *(const v8bf*)gsrc;   // sync fallback through VGPRs
#endif
}

__device__ __forceinline__ void async_wait0() {
#if EVD_HAS_ASYNC_LDS
  #if __has_builtin(__builtin_amdgcn_s_wait_asynccnt)
    __builtin_amdgcn_s_wait_asynccnt(0);
  #else
    asm volatile("s_wait_asynccnt 0x0" ::: "memory");
  #endif
#endif
}

// ---------- WMMA fragment loads (bf16, ISA 16-bit layouts) ----------
// A 16x32: lane half 0 -> K {0..7} and {16..23}; half 1 -> +8.
__device__ __forceinline__ v16bf load_fragA(const __bf16* rowPtr, int half) {
    v8bf r1 = *(const v8bf*)(rowPtr + 8 * half);
    v8bf r2 = *(const v8bf*)(rowPtr + 16 + 8 * half);
    return __builtin_shufflevector(r1, r2, 0,1,2,3,4,5,6,7,8,9,10,11,12,13,14,15);
}
// B 32x16: lane half 0 -> K {0..7} and {16..23}; half 1 -> +8.
__device__ __forceinline__ v16bf load_fragB(const __bf16* rowPtr, int half) {
    v8bf r1 = *(const v8bf*)(rowPtr + 8 * half);
    v8bf r2 = *(const v8bf*)(rowPtr + 16 + 8 * half);
    return __builtin_shufflevector(r1, r2, 0,1,2,3,4,5,6,7,8,9,10,11,12,13,14,15);
}

// ---------- fused epilogue (shared) ----------
__device__ __forceinline__ void epilogue_and_reduce(v8f c0, v8f c1, v8f c2, v8f c3,
                                                    int p0, int q0, int wave, int half,
                                                    int l16, int lane, float* out) {
    const float CPOS = 2.0f / (2048.0f * 12.0f);
    const float CNEG = 1.0f / (2048.0f * 2047.0f * 16.0f);
    float lsum = 0.0f;
    const int prow_base = p0 + wave * 16 + half * 8;   // C/D: M = vgpr + 8*(lane>=16)
    #pragma unroll
    for (int nsub = 0; nsub < 4; ++nsub) {
        v8f c = (nsub == 0) ? c0 : (nsub == 1) ? c1 : (nsub == 2) ? c2 : c3;
        const int q = q0 + nsub * 16 + l16;
        #pragma unroll
        for (int m = 0; m < 8; ++m) {
            float g = c[m];
            float t    = fmaxf(g, __shfl_xor(g, 1, 32));
            float gmax = fmaxf(t, __shfl_xor(t, 2, 32));
            float e = __expf(g - gmax);
            int p = prow_base + m;
            if ((p >> 2) == (q >> 2)) {
                if ((p & 3) != (q & 3)) lsum -= CPOS * e;   // positive pairs (i==k, j!=l)
            } else {
                lsum += CNEG * fmaxf(e * e, 1e-8f);         // negative pairs (i!=k)
            }
        }
    }
    #pragma unroll
    for (int off = 16; off > 0; off >>= 1)
        lsum += __shfl_xor(lsum, off, 32);
    if (lane == 0)
        atomicAdd(out, lsum);
}

// ================= fast path: pre-split bf16 + async double-buffered LDS =================
__launch_bounds__(128)
__global__ void evd_gram_fast(const __bf16* __restrict__ Whi, const __bf16* __restrict__ Wlo,
                              float* __restrict__ out)
{
    constexpr int D      = 256;
    constexpr int KC     = 32;              // one WMMA K-slice per chunk
    constexpr int STRIDE = 40;              // bf16 row stride (80 B)
    constexpr int PANEL  = 64 * STRIDE;     // bf16 elems per sub-panel

    __shared__ __bf16 lds[2 * 4 * PANEL];   // [buf][Ahi,Alo,Bhi,Blo]

    const int tid  = threadIdx.x;
    const int lane = tid & 31;
    const int wave = tid >> 5;
    const int p0   = blockIdx.y * 64;
    const int q0   = blockIdx.x * 64;
    const int half = lane >> 4;
    const int l16  = lane & 15;

    v8f c0 = {}, c1 = {}, c2 = {}, c3 = {};

    auto issue = [&](int buf, int kc) {
        __bf16* base = &lds[buf * 4 * PANEL];
        #pragma unroll
        for (int v = 0; v < 2; ++v) {
            int idx = tid + v * 128;        // 0..255 b128 slots per sub-panel
            int row = idx >> 2;
            int c8  = (idx & 3) * 8;        // bf16 units, 16B aligned
            size_t gA = (size_t)(p0 + row) * D + kc + c8;
            size_t gB = (size_t)(q0 + row) * D + kc + c8;
            int loff  = row * STRIDE + c8;
            async_copy16(Whi + gA, base + 0 * PANEL + loff);
            async_copy16(Wlo + gA, base + 1 * PANEL + loff);
            async_copy16(Whi + gB, base + 2 * PANEL + loff);
            async_copy16(Wlo + gB, base + 3 * PANEL + loff);
        }
    };

    auto compute = [&](int buf) {
        const __bf16* base = &lds[buf * 4 * PANEL];
        v16bf aHi = load_fragA(base + 0 * PANEL + (wave * 16 + l16) * STRIDE, half);
        v16bf aLo = load_fragA(base + 1 * PANEL + (wave * 16 + l16) * STRIDE, half);
        #define EVD_STEP(CC, NS)                                                          \
        {                                                                                 \
            v16bf bHi = load_fragB(base + 2 * PANEL + ((NS) * 16 + l16) * STRIDE, half);  \
            v16bf bLo = load_fragB(base + 3 * PANEL + ((NS) * 16 + l16) * STRIDE, half);  \
            CC = __builtin_amdgcn_wmma_f32_16x16x32_bf16(false, aHi, false, bHi,          \
                                                         (short)0, CC, false, false);     \
            CC = __builtin_amdgcn_wmma_f32_16x16x32_bf16(false, aHi, false, bLo,          \
                                                         (short)0, CC, false, false);     \
            CC = __builtin_amdgcn_wmma_f32_16x16x32_bf16(false, aLo, false, bHi,          \
                                                         (short)0, CC, false, false);     \
        }
        EVD_STEP(c0, 0) EVD_STEP(c1, 1) EVD_STEP(c2, 2) EVD_STEP(c3, 3)
        #undef EVD_STEP
    };

    // software pipeline: prefetch chunk c+1 while computing chunk c
    issue(0, 0);
    async_wait0();
    __syncthreads();
    for (int c = 0; c < D / KC; ++c) {
        int cur = c & 1;
        if (c + 1 < D / KC) issue(cur ^ 1, (c + 1) * KC);
        compute(cur);
        async_wait0();
        __syncthreads();
    }

    epilogue_and_reduce(c0, c1, c2, c3, p0, q0, wave, half, l16, lane, out);
}

// ================= fallback: self-converting (no workspace) =================
__launch_bounds__(128)
__global__ void evd_gram_fallback(const float* __restrict__ A, float* __restrict__ out)
{
    constexpr int D      = 256;
    constexpr int KC     = 64;
    constexpr int STRIDE = 72;

    __shared__ __bf16 ldsAhi[64 * STRIDE];
    __shared__ __bf16 ldsAlo[64 * STRIDE];
    __shared__ __bf16 ldsBhi[64 * STRIDE];
    __shared__ __bf16 ldsBlo[64 * STRIDE];

    const int tid  = threadIdx.x;
    const int lane = tid & 31;
    const int wave = tid >> 5;
    const int p0   = blockIdx.y * 64;
    const int q0   = blockIdx.x * 64;
    const int half = lane >> 4;
    const int l16  = lane & 15;

    v8f c0 = {}, c1 = {}, c2 = {}, c3 = {};

    for (int kc = 0; kc < D; kc += KC) {
        __syncthreads();
        float4 ra[8], rb[8];
        #pragma unroll
        for (int v = 0; v < 8; ++v) {
            int idx = tid + v * 128;
            int row = idx >> 4;
            int c4  = (idx & 15) << 2;
            ra[v] = *(const float4*)&A[(size_t)(p0 + row) * D + kc + c4];
            rb[v] = *(const float4*)&A[(size_t)(q0 + row) * D + kc + c4];
        }
        #pragma unroll
        for (int v = 0; v < 8; ++v) {
            int idx = tid + v * 128;
            int row = idx >> 4;
            int c4  = (idx & 15) << 2;
            v4bf hi, lo;
            split4(ra[v], &hi, &lo);
            *(v4bf*)&ldsAhi[row * STRIDE + c4] = hi;
            *(v4bf*)&ldsAlo[row * STRIDE + c4] = lo;
            split4(rb[v], &hi, &lo);
            *(v4bf*)&ldsBhi[row * STRIDE + c4] = hi;
            *(v4bf*)&ldsBlo[row * STRIDE + c4] = lo;
        }
        __syncthreads();

        #pragma unroll
        for (int ks = 0; ks < KC; ks += 32) {
            v16bf aHi = load_fragA(&ldsAhi[(wave * 16 + l16) * STRIDE + ks], half);
            v16bf aLo = load_fragA(&ldsAlo[(wave * 16 + l16) * STRIDE + ks], half);
            #define EVD_STEP(CC, NS)                                                      \
            {                                                                             \
                v16bf bHi = load_fragB(&ldsBhi[((NS) * 16 + l16) * STRIDE + ks], half);   \
                v16bf bLo = load_fragB(&ldsBlo[((NS) * 16 + l16) * STRIDE + ks], half);   \
                CC = __builtin_amdgcn_wmma_f32_16x16x32_bf16(false, aHi, false, bHi,      \
                                                             (short)0, CC, false, false); \
                CC = __builtin_amdgcn_wmma_f32_16x16x32_bf16(false, aHi, false, bLo,      \
                                                             (short)0, CC, false, false); \
                CC = __builtin_amdgcn_wmma_f32_16x16x32_bf16(false, aLo, false, bHi,      \
                                                             (short)0, CC, false, false); \
            }
            EVD_STEP(c0, 0) EVD_STEP(c1, 1) EVD_STEP(c2, 2) EVD_STEP(c3, 3)
            #undef EVD_STEP
        }
    }

    epilogue_and_reduce(c0, c1, c2, c3, p0, q0, wave, half, l16, lane, out);
}

extern "C" void kernel_launch(void* const* d_in, const int* in_sizes, int n_in,
                              void* d_out, int out_size, void* d_ws, size_t ws_size,
                              hipStream_t stream) {
    (void)in_sizes; (void)n_in; (void)out_size;
    const float* a = (const float*)d_in[0];
    float* out = (float*)d_out;

    constexpr size_t ELEMS = 8192u * 256u;          // 2M elements
    constexpr size_t NEED  = ELEMS * 2u * 2u;       // Whi + Wlo bf16 = 8 MB

    evd_zero_out<<<dim3(1), dim3(1), 0, stream>>>(out);

    dim3 grid(128, 128);                            // 8192/64 supertiles per side
    if (ws_size >= NEED) {
        __bf16* whi = (__bf16*)d_ws;
        __bf16* wlo = whi + ELEMS;
        evd_split_bf16<<<dim3(2048), dim3(256), 0, stream>>>(a, whi, wlo);
        evd_gram_fast<<<grid, dim3(128), 0, stream>>>(whi, wlo, out);
    } else {
        evd_gram_fallback<<<grid, dim3(128), 0, stream>>>(a, out);
    }
}
